// RelativeMultiHeadAttention_71777493451214
// MI455X (gfx1250) — matrix operand order repaired and checked by reading, not compiled, not run on hardware
//
#include <hip/hip_runtime.h>
#include <hip/hip_bf16.h>
#include <stdint.h>

#define S_LEN 2048
#define C_DIM 512
#define H_NUM 16
#define D_DIM 32
#define NROWS 4096            // B*S
#define PSTR 2056             // panel row stride (bf16 elems), padded vs 2048 for banks
#define PBSTR 40              // prob-bounce row stride (bf16 elems), 80B => 16B-aligned b128

typedef __attribute__((ext_vector_type(16))) __bf16 v16bf;
typedef __attribute__((ext_vector_type(8)))  float  v8f;

union BF16x16 { v16bf v; uint4 q[2]; unsigned short s[16]; };

__device__ __forceinline__ unsigned short f2bf(float f) {
  union { float f; uint32_t u; } x; x.f = f;
  uint32_t r = x.u + 0x7fffu + ((x.u >> 16) & 1u);   // round-to-nearest-even
  return (unsigned short)(r >> 16);
}
__device__ __forceinline__ float bf2f(unsigned short b) {
  union { uint32_t u; float f; } x; x.u = ((uint32_t)b) << 16;
  return x.f;
}

// ---------------- fp32 -> bf16 convert ----------------
__global__ void cvt_kernel(const float* __restrict__ src,
                           unsigned short* __restrict__ dst, int n) {
  int i = blockIdx.x * blockDim.x + threadIdx.x;
  int stride = gridDim.x * blockDim.x;
  for (; i < n; i += stride) dst[i] = f2bf(src[i]);
}

// ---------------- projection GEMM: Y = X * W^T (+bias), WMMA bf16 ----------------
// mode 0: Q -> out0 = Y+bias+ubias (bf16), out1 = Y+bias+vbias (bf16)
// mode 1: plain bf16 out0 = Y+bias (bias may be null)
// mode 2: V -> outT transposed [B,H,D,S] bf16, += bias
// mode 3: final fp32 outF = Y+bias
__global__ __launch_bounds__(256) void proj_kernel(
    const unsigned short* __restrict__ X, const unsigned short* __restrict__ W,
    const float* __restrict__ bias, const float* __restrict__ ubias,
    const float* __restrict__ vbias,
    unsigned short* __restrict__ out0, unsigned short* __restrict__ out1,
    unsigned short* __restrict__ outT, float* __restrict__ outF, int mode) {
  const int lane = threadIdx.x & 31;
  const int w    = threadIdx.x >> 5;
  const int hf   = lane >> 4;
  const int l15  = lane & 15;
  const int rowbase = blockIdx.x * 32 + (w >> 2) * 16;
  const int colbase = blockIdx.y * 64 + (w & 3) * 16;
  const int arow = rowbase + l15;
  const int bcol = colbase + l15;

  v8f acc = {};
  for (int k0 = 0; k0 < C_DIM; k0 += 32) {
    BF16x16 a, b;
    const unsigned short* ap = X + (size_t)arow * C_DIM + k0 + hf * 16;
    a.q[0] = *(const uint4*)(ap);
    a.q[1] = *(const uint4*)(ap + 8);
    const unsigned short* bp = W + (size_t)bcol * C_DIM + k0 + hf * 16;
    b.q[0] = *(const uint4*)(bp);
    b.q[1] = *(const uint4*)(bp + 8);
    acc = __builtin_amdgcn_wmma_f32_16x16x32_bf16(false, a.v, false, b.v,
                                                  (short)0, acc, false, false);
  }

  const float bb  = bias  ? bias[bcol]  : 0.0f;
  const float ubb = (mode == 0) ? ubias[bcol] : 0.0f;
  const float vbb = (mode == 0) ? vbias[bcol] : 0.0f;

#pragma unroll
  for (int v = 0; v < 8; ++v) {
    const int row = rowbase + v + 8 * hf;   // D layout: M = vgpr + 8*(lane>=16)
    const int col = bcol;                   // N = lane&15
    const float y = acc[v] + bb;
    if (mode == 0) {
      out0[(size_t)row * C_DIM + col] = f2bf(y + ubb);
      out1[(size_t)row * C_DIM + col] = f2bf(y + vbb);
    } else if (mode == 1) {
      out0[(size_t)row * C_DIM + col] = f2bf(y);
    } else if (mode == 2) {
      const int b_  = row >> 11, s = row & (S_LEN - 1);
      const int h   = col >> 5,  d = col & (D_DIM - 1);
      outT[(((size_t)b_ * H_NUM + h) * D_DIM + d) * S_LEN + s] = f2bf(y);
    } else {
      outF[(size_t)row * C_DIM + col] = y;
    }
  }
}

// ---------------- fused relative attention ----------------
// grid: (S/16 query blocks, B*H). 8 waves: all share the 16-row query tile,
// split the 128 key tiles. Pos skew handled via an LDS panel of pos-scores.
__global__ __launch_bounds__(256) void attn_kernel(
    const unsigned short* __restrict__ qu, const unsigned short* __restrict__ qv,
    const unsigned short* __restrict__ kb, const unsigned short* __restrict__ pb,
    const unsigned short* __restrict__ vt, unsigned short* __restrict__ ctx) {
  __shared__ unsigned short panel[17 * PSTR];       // pos-score rows i0..i0+16, bf16
  __shared__ unsigned short pbuf[8 * 16 * PBSTR];   // per-wave prob bounce
  __shared__ float cmb[8 * 16 * 33];                // per-wave context partials
  __shared__ float smax[8 * 16];
  __shared__ float ssum[8 * 16];

  const int tid  = threadIdx.x;
  const int lane = tid & 31, w = tid >> 5, hf = lane >> 4, l15 = lane & 15;
  const int i0 = blockIdx.x * 16;
  const int bh = blockIdx.y;
  const size_t base = ((size_t)(bh >> 4) * S_LEN) * C_DIM + (bh & 15) * D_DIM;
  const float scale = 0.04419417382415922f;   // 1/sqrt(512)

  // -------- phase 1: pos panel = Qv_block x Pos^T  (WMMA) --------
  for (int t = w; t < 256; t += 8) {
    const int rt = t >> 7, ct = t & 127;
    int arow = i0 + rt * 16 + l15; if (arow >= S_LEN) arow = S_LEN - 1;
    BF16x16 a, b;
    const unsigned short* ap = qv + base + (size_t)arow * C_DIM + hf * 16;
    a.q[0] = *(const uint4*)ap;  a.q[1] = *(const uint4*)(ap + 8);
    const unsigned short* bp = pb + base + (size_t)(ct * 16 + l15) * C_DIM + hf * 16;
    b.q[0] = *(const uint4*)bp;  b.q[1] = *(const uint4*)(bp + 8);
    v8f c = {};
    c = __builtin_amdgcn_wmma_f32_16x16x32_bf16(false, a.v, false, b.v,
                                                (short)0, c, false, false);
#pragma unroll
    for (int v = 0; v < 8; ++v) {
      const int pr = rt * 16 + v + 8 * hf;
      if (pr < 17) panel[pr * PSTR + ct * 16 + l15] = f2bf(c[v]);
    }
  }
  __syncthreads();

  // -------- phase 2: flash loop over key tiles --------
  BF16x16 au;
  {
    const unsigned short* ap = qu + base + (size_t)(i0 + l15) * C_DIM + hf * 16;
    au.q[0] = *(const uint4*)ap;  au.q[1] = *(const uint4*)(ap + 8);
  }
  float rowmax[8], rowsum[8];
#pragma unroll
  for (int v = 0; v < 8; ++v) { rowmax[v] = -1e30f; rowsum[v] = 0.0f; }
  v8f acc0 = {}, acc1 = {};

  for (int t = 0; t < 8; ++t) {
    const int jtA = 2 * (w + 8 * t), jtB = jtA + 1;
    v8f csA, csB;
    {
      BF16x16 bk;
      const unsigned short* bp = kb + base + (size_t)(jtA * 16 + l15) * C_DIM + hf * 16;
      bk.q[0] = *(const uint4*)bp;  bk.q[1] = *(const uint4*)(bp + 8);
      v8f z = {};
      csA = __builtin_amdgcn_wmma_f32_16x16x32_bf16(false, au.v, false, bk.v,
                                                    (short)0, z, false, false);
      bp = kb + base + (size_t)(jtB * 16 + l15) * C_DIM + hf * 16;
      bk.q[0] = *(const uint4*)bp;  bk.q[1] = *(const uint4*)(bp + 8);
      csB = __builtin_amdgcn_wmma_f32_16x16x32_bf16(false, au.v, false, bk.v,
                                                    (short)0, z, false, false);
    }
    // skew gather from panel + scale
#pragma unroll
    for (int v = 0; v < 8; ++v) {
      const int m = v + 8 * hf;
      const int i = i0 + m;
      {
        const int j = jtA * 16 + l15;
        const bool past = (j <= i);
        const int pr = past ? m : (m + 1);
        int pc = past ? (j - i + S_LEN - 1) : (j - i - 2);
        const bool zr = (j == i + 1);
        if (pc < 0) pc = 0;
        float pv = bf2f(panel[pr * PSTR + pc]);
        if (zr) pv = 0.0f;
        csA[v] = (csA[v] + pv) * scale;
      }
      {
        const int j = jtB * 16 + l15;
        const bool past = (j <= i);
        const int pr = past ? m : (m + 1);
        int pc = past ? (j - i + S_LEN - 1) : (j - i - 2);
        const bool zr = (j == i + 1);
        if (pc < 0) pc = 0;
        float pv = bf2f(panel[pr * PSTR + pc]);
        if (zr) pv = 0.0f;
        csB[v] = (csB[v] + pv) * scale;
      }
    }
    // online softmax (row reductions across the 16-lane half)
#pragma unroll
    for (int v = 0; v < 8; ++v) {
      float mt = fmaxf(csA[v], csB[v]);
      mt = fmaxf(mt, __shfl_xor(mt, 1));
      mt = fmaxf(mt, __shfl_xor(mt, 2));
      mt = fmaxf(mt, __shfl_xor(mt, 4));
      mt = fmaxf(mt, __shfl_xor(mt, 8));
      const float nm = fmaxf(rowmax[v], mt);
      const float corr = __expf(rowmax[v] - nm);
      rowmax[v] = nm;
      const float pA = __expf(csA[v] - nm);
      const float pB = __expf(csB[v] - nm);
      float ps = pA + pB;
      ps += __shfl_xor(ps, 1);
      ps += __shfl_xor(ps, 2);
      ps += __shfl_xor(ps, 4);
      ps += __shfl_xor(ps, 8);
      rowsum[v] = rowsum[v] * corr + ps;
      acc0[v] *= corr;  acc1[v] *= corr;
      const int m = v + 8 * hf;
      pbuf[(w * 16 + m) * PBSTR + l15]      = f2bf(pA);
      pbuf[(w * 16 + m) * PBSTR + 16 + l15] = f2bf(pB);
    }
    asm volatile("s_wait_dscnt 0" ::: "memory");   // stores visible before reload
    // reload probs in A layout, V in B layout (transposed V => contiguous)
    BF16x16 apv, bv;
    {
      const int rb = (w * 16 + l15) * PBSTR;
      apv.q[0] = *(const uint4*)&pbuf[rb + hf * 16];
      apv.q[1] = *(const uint4*)&pbuf[rb + hf * 16 + 8];
    }
    const size_t vtb = (size_t)bh * D_DIM * S_LEN;
    const int tok16 = (hf ? jtB : jtA) * 16;
    const unsigned short* vp = vt + vtb + (size_t)(l15) * S_LEN + tok16;
    bv.q[0] = *(const uint4*)vp;  bv.q[1] = *(const uint4*)(vp + 8);
    acc0 = __builtin_amdgcn_wmma_f32_16x16x32_bf16(false, apv.v, false, bv.v,
                                                   (short)0, acc0, false, false);
    vp = vt + vtb + (size_t)(16 + l15) * S_LEN + tok16;
    bv.q[0] = *(const uint4*)vp;  bv.q[1] = *(const uint4*)(vp + 8);
    acc1 = __builtin_amdgcn_wmma_f32_16x16x32_bf16(false, apv.v, false, bv.v,
                                                   (short)0, acc1, false, false);
  }

  // -------- phase 3: cross-wave softmax merge --------
#pragma unroll
  for (int v = 0; v < 8; ++v) {
    const int m = v + 8 * hf;
    cmb[(w * 16 + m) * 33 + l15]      = acc0[v];
    cmb[(w * 16 + m) * 33 + 16 + l15] = acc1[v];
    if (l15 == 0) { smax[w * 16 + m] = rowmax[v]; ssum[w * 16 + m] = rowsum[v]; }
  }
  __syncthreads();
  {
    const int m = tid >> 4, d = tid & 15;
    float M = -1e30f;
#pragma unroll
    for (int ww = 0; ww < 8; ++ww) M = fmaxf(M, smax[ww * 16 + m]);
    float L = 0.0f, c0 = 0.0f, c1 = 0.0f;
#pragma unroll
    for (int ww = 0; ww < 8; ++ww) {
      const float e = __expf(smax[ww * 16 + m] - M);
      L  += e * ssum[ww * 16 + m];
      c0 += e * cmb[(ww * 16 + m) * 33 + d];
      c1 += e * cmb[(ww * 16 + m) * 33 + 16 + d];
    }
    const float inv = 1.0f / L;
    ctx[base + (size_t)(i0 + m) * C_DIM + d]      = f2bf(c0 * inv);
    ctx[base + (size_t)(i0 + m) * C_DIM + 16 + d] = f2bf(c1 * inv);
  }
}

extern "C" void kernel_launch(void* const* d_in, const int* in_sizes, int n_in,
                              void* d_out, int out_size, void* d_ws, size_t ws_size,
                              hipStream_t stream) {
  (void)in_sizes; (void)n_in; (void)out_size; (void)ws_size;
  const float* q  = (const float*)d_in[0];
  const float* k  = (const float*)d_in[1];
  const float* v  = (const float*)d_in[2];
  const float* pe = (const float*)d_in[3];
  const float* Wq = (const float*)d_in[4];
  const float* bq = (const float*)d_in[5];
  const float* Wk = (const float*)d_in[6];
  const float* bk = (const float*)d_in[7];
  const float* Wv = (const float*)d_in[8];
  const float* bv = (const float*)d_in[9];
  const float* Wp = (const float*)d_in[10];
  const float* ub = (const float*)d_in[11];
  const float* vb = (const float*)d_in[12];
  const float* Wo = (const float*)d_in[13];
  const float* bo = (const float*)d_in[14];

  const size_t NSC = (size_t)2 * S_LEN * C_DIM;   // 2,097,152
  const size_t WSZ = (size_t)C_DIM * C_DIM;       // 262,144
  unsigned short* p = (unsigned short*)d_ws;
  unsigned short* xq = p; p += NSC;
  unsigned short* xk = p; p += NSC;
  unsigned short* xv = p; p += NSC;
  unsigned short* xp = p; p += NSC;
  unsigned short* wq = p; p += WSZ;
  unsigned short* wk = p; p += WSZ;
  unsigned short* wv = p; p += WSZ;
  unsigned short* wp = p; p += WSZ;
  unsigned short* wo = p; p += WSZ;
  unsigned short* quB = p; p += NSC;
  unsigned short* qvB = p; p += NSC;
  unsigned short* kB  = p; p += NSC;
  unsigned short* pB  = p; p += NSC;
  unsigned short* vtB = p; p += NSC;
  unsigned short* ctx = p; p += NSC;

  cvt_kernel<<<1024, 256, 0, stream>>>(q,  xq, (int)NSC);
  cvt_kernel<<<1024, 256, 0, stream>>>(k,  xk, (int)NSC);
  cvt_kernel<<<1024, 256, 0, stream>>>(v,  xv, (int)NSC);
  cvt_kernel<<<1024, 256, 0, stream>>>(pe, xp, (int)NSC);
  cvt_kernel<<<256, 256, 0, stream>>>(Wq, wq, (int)WSZ);
  cvt_kernel<<<256, 256, 0, stream>>>(Wk, wk, (int)WSZ);
  cvt_kernel<<<256, 256, 0, stream>>>(Wv, wv, (int)WSZ);
  cvt_kernel<<<256, 256, 0, stream>>>(Wp, wp, (int)WSZ);
  cvt_kernel<<<256, 256, 0, stream>>>(Wo, wo, (int)WSZ);

  dim3 pg(NROWS / 32, C_DIM / 64);   // 128 x 8
  proj_kernel<<<pg, 256, 0, stream>>>(xq, wq, bq, ub, vb, quB, qvB, nullptr, nullptr, 0);
  proj_kernel<<<pg, 256, 0, stream>>>(xk, wk, bk, nullptr, nullptr, kB, nullptr, nullptr, nullptr, 1);
  proj_kernel<<<pg, 256, 0, stream>>>(xv, wv, bv, nullptr, nullptr, nullptr, nullptr, vtB, nullptr, 2);
  proj_kernel<<<pg, 256, 0, stream>>>(xp, wp, nullptr, nullptr, nullptr, pB, nullptr, nullptr, nullptr, 1);

  attn_kernel<<<dim3(S_LEN / 16, 2 * H_NUM), 256, 0, stream>>>(quB, qvB, kB, pB, vtB, ctx);

  proj_kernel<<<pg, 256, 0, stream>>>(ctx, wo, bo, nullptr, nullptr, nullptr, nullptr, nullptr,
                                      (float*)d_out, 3);
}
